// GCUMotion_43997644980922
// MI455X (gfx1250) — matrix operand order repaired and checked by reading, not compiled, not run on hardware
//
#include <hip/hip_runtime.h>

// ---------------------------------------------------------------------------
// GCUMotion edge-conv network for MI455X (gfx1250, wave32, WMMA).
//  N=25000 nodes, E=400000 edges (+N self loops), C_IN=64, C_OUT=256.
// Layer-1 of each edge MLP is factored into per-node GEMMs; layer-2 is a
// fused per-edge bf16 WMMA GEMM whose epilogue does ReLU + BN-stat
// accumulation + segment-max via integer atomic-max (valid for >=0 floats).
// This revision adds block-uniform full-tile fast paths in all GEMM
// epilogues to eliminate per-element exec-mask branching.
// ---------------------------------------------------------------------------

#define NN   25000
#define EE   400000
#define ENTOT 425000      // EE + NN self loops
#define CIN  64
#define CH   128          // C_OUT/2
#define PF   16
#define DCAT 288
#define CO   256
#define BNEPS 1e-5f

typedef __attribute__((ext_vector_type(16))) __bf16 v16bf;
typedef __attribute__((ext_vector_type(8)))  float  v8f;

// per-graph stat-block layout (floats)
enum { SUM1X=0, SQ1X=128, SUM1P=256, SQ1P=272, S1X=288, T1X=416,
       S1P=544, T1P=560, SUM2X=576, SQ2X=704, SUM2P=832, SQ2P=848, STATSZ=1024 };
// final BN stat block
enum { SUMZ=0, SQZ=256, AOUT=512, COUTO=768, FSTSZ=1024 };

__device__ __forceinline__ unsigned short f2bf(float f) {
  unsigned int u = __float_as_uint(f);
  u += 0x7fffu + ((u >> 16) & 1u);           // round-to-nearest-even
  return (unsigned short)(u >> 16);
}

// LDS index (in bf16 units) for A-matrix fragment element (row, ch),
// contiguous-k A layout (matches B packing): lane = row + 16*k[4], dword = k[3:1].
// Each lane's 16 bf16 are contiguous (32B) so a lane loads its fragment
// as one aligned v16bf read.
__device__ __forceinline__ int afrag_idx(int row, int ch) {
  int kstep = ch >> 5;
  int lane  = (row & 15) + (((ch >> 4) & 1) << 4);
  int d     = (ch >> 1) & 7;
  return ((kstep * 32 + lane) << 4) + (d << 1) + (ch & 1);
}

// ---------------------------------------------------------------------------
__global__ void k_zero(float* p, int n) {
  int i = blockIdx.x * 256 + threadIdx.x;
  if (i < n) p[i] = 0.f;
}

__global__ void k_cvt_bf16(const float* x, unsigned short* o, int n) {
  int i = blockIdx.x * 256 + threadIdx.x;
  if (i < n) o[i] = f2bf(x[i]);
}

// Pack W1 (128x128 row-major, [k][n]) into two B-fragment streams:
//  fragD = (W1_top - W1_bot), fragB = W1_bot  (both 64xK x 128 cols).
// B fragment order: dword idx = ((tile*KS + kstep)*32 + lane)*8 + d,
//  col = tile*16 + lane[3:0], k = kstep*32 + lane[4]*16 + d*2 + {0,1}.
__global__ void k_pack_w1(const float* W1, unsigned int* fragD, unsigned int* fragB) {
  int id = blockIdx.x * 256 + threadIdx.x;
  if (id >= 8 * 2 * 32 * 8) return;
  int d = id & 7, l = (id >> 3) & 31, rest = id >> 8;
  int s = rest & 1, t = rest >> 1;
  int col = t * 16 + (l & 15);
  int k   = s * 32 + ((l >> 4) << 4) + (d << 1);
  float t0 = W1[k * CH + col],        t1 = W1[(k + 1) * CH + col];
  float b0 = W1[(64 + k) * CH + col], b1 = W1[(64 + k + 1) * CH + col];
  fragB[id] = ((unsigned)f2bf(b1) << 16) | f2bf(b0);
  fragD[id] = ((unsigned)f2bf(t1 - b1) << 16) | f2bf(t0 - b0);
}

// Generic B-fragment packer for a KxNC row-major f32 weight matrix.
__global__ void k_pack_gen(const float* W, unsigned int* frag,
                           int K, int NC, int ntiles, int ksteps) {
  int id = blockIdx.x * 256 + threadIdx.x;
  if (id >= ntiles * ksteps * 32 * 8) return;
  int d = id & 7, l = (id >> 3) & 31, rest = id >> 8;
  int s = rest % ksteps, t = rest / ksteps;
  int col = t * 16 + (l & 15);
  int k   = s * 32 + ((l >> 4) << 4) + (d << 1);
  float w0 = (k     < K) ? W[(size_t)k * NC + col]       : 0.f;
  float w1 = (k + 1 < K) ? W[(size_t)(k + 1) * NC + col] : 0.f;
  frag[id] = ((unsigned)f2bf(w1) << 16) | f2bf(w0);
}

// out[N x 128] = x_bf16[N x 64] @ Wfrag (pre-packed).  16-row tiles, 8 waves.
__global__ void k_node_gemm(const unsigned short* xbf, const unsigned int* wfrag,
                            float* out) {
  __shared__ __align__(32) unsigned short af[2 * 32 * 16];
  int tid = threadIdx.x, base = blockIdx.x * 16;
  for (int v = tid; v < 16 * CIN; v += 256) {
    int row = v >> 6, ch = v & 63;
    int node = base + row;
    af[afrag_idx(row, ch)] = (node < NN) ? xbf[node * CIN + ch] : (unsigned short)0;
  }
  __syncthreads();
  int w = tid >> 5, lane = tid & 31;
  v8f acc = {0.f, 0.f, 0.f, 0.f, 0.f, 0.f, 0.f, 0.f};
  #pragma unroll
  for (int s = 0; s < 2; s++) {
    v16bf a = *reinterpret_cast<const v16bf*>(&af[(s * 32 + lane) * 16]);
    v16bf b = *reinterpret_cast<const v16bf*>(wfrag + (size_t)((w * 2 + s) * 32 + lane) * 8);
    acc = __builtin_amdgcn_wmma_f32_16x16x32_bf16(false, a, false, b, (short)0, acc, false, false);
  }
  int col  = w * 16 + (lane & 15);
  int row0 = base + ((lane >> 4) << 3);
  float* p = out + (size_t)row0 * CH + col;
  if (base + 16 <= NN) {                       // full tile: straight-line stores
    #pragma unroll
    for (int r = 0; r < 8; r++) p[(size_t)r * CH] = acc[r];
  } else {
    #pragma unroll
    for (int r = 0; r < 8; r++)
      if (row0 + r < NN) p[(size_t)r * CH] = acc[r];
  }
}

// PA = pos @ (Wp1_top - Wp1_bot), PB = pos @ Wp1_bot   (6->16 split 3/3)
__global__ void k_pos_pre(const float* pos, const float* Wp1, float* PA, float* PB) {
  int id = blockIdx.x * 256 + threadIdx.x;
  if (id >= NN * PF) return;
  int n = id >> 4, c = id & 15;
  float a = 0.f, b = 0.f;
  for (int d = 0; d < 3; d++) {
    float pv = pos[n * 3 + d];
    float wt = Wp1[d * PF + c], wb = Wp1[(3 + d) * PF + c];
    a += pv * (wt - wb);
    b += pv * wb;
  }
  PA[id] = a; PB[id] = b;
}

// Pass 1: stats of relu(layer1) over all EN edges, x-path (128 ch) + pos (16 ch).
__global__ void k_stats1(const int* eidx, const float* Ax, const float* Bx,
                         const float* b1x, const float* PA, const float* PB,
                         const float* b1p, float* st) {
  int tid = threadIdx.x, base = blockIdx.x * 64;
  { // x path: 2 edges in flight, channel = tid&127
    int c = tid & 127, rh = tid >> 7;
    float bb = b1x[c], s = 0.f, q = 0.f;
    for (int i = 0; i < 32; i++) {
      int e = base + rh + (i << 1);
      if (e >= ENTOT) break;
      int sj, di;
      if (e < EE) { sj = eidx[e]; di = eidx[EE + e]; } else { sj = di = e - EE; }
      float v = Ax[(size_t)di * CH + c] + Bx[(size_t)sj * CH + c] + bb;
      v = fmaxf(v, 0.f);
      s += v; q += v * v;
    }
    if (s != 0.f || q != 0.f) { atomicAdd(&st[SUM1X + c], s); atomicAdd(&st[SQ1X + c], q); }
  }
  { // pos path: 16 edges in flight, channel = tid&15
    int c = tid & 15, row = tid >> 4;
    float bb = b1p[c], s = 0.f, q = 0.f;
    for (int i = 0; i < 4; i++) {
      int e = base + row + (i << 4);
      if (e >= ENTOT) break;
      int sj, di;
      if (e < EE) { sj = eidx[e]; di = eidx[EE + e]; } else { sj = di = e - EE; }
      float v = PA[di * PF + c] + PB[sj * PF + c] + bb;
      v = fmaxf(v, 0.f);
      s += v; q += v * v;
    }
    if (s != 0.f || q != 0.f) { atomicAdd(&st[SUM1P + c], s); atomicAdd(&st[SQ1P + c], q); }
  }
}

// Finalize BN1: store affine n = s1*relu(h)+t1 per channel.
__global__ void k_fin_bn1(float* st, const float* g1x, const float* be1x,
                          const float* g1p, const float* be1p) {
  int c = threadIdx.x;
  const float inv_n = 1.f / (float)ENTOT;
  if (c < CH) {
    float m = st[SUM1X + c] * inv_n;
    float v = fmaxf(st[SQ1X + c] * inv_n - m * m, 0.f);
    float a = g1x[c] * rsqrtf(v + BNEPS);
    st[S1X + c] = a; st[T1X + c] = be1x[c] - a * m;
  } else if (c < CH + PF) {
    int cc = c - CH;
    float m = st[SUM1P + cc] * inv_n;
    float v = fmaxf(st[SQ1P + cc] * inv_n - m * m, 0.f);
    float a = g1p[cc] * rsqrtf(v + BNEPS);
    st[S1P + cc] = a; st[T1P + cc] = be1p[cc] - a * m;
  }
}

// Workhorse: per-edge-tile fused  n1 = bn(relu(Ax[dst]+Bx[src]+b1)),
// h2 = n1 @ W2 (bf16 WMMA), r2 = relu(h2+b2); epilogue: BN2 stats + atomic
// segment-max into maxx.  16 edges x 128 cols per workgroup (8 waves).
__global__ void k_edge_gemm(const int* eidx, const float* Ax, const float* Bx,
                            const float* b1x, const float* b2x,
                            const unsigned int* w2f, float* st, float* maxx) {
  __shared__ __align__(32) unsigned short nf[4 * 32 * 16];
  __shared__ int sdst[16], ssrc[16];
  __shared__ float sB1[CH], sS1[CH], sT1[CH];
  int tid = threadIdx.x, base = blockIdx.x * 16;
  if (tid < 16) {
    int e = base + tid, sj, di;
    if (e < EE)         { sj = eidx[e]; di = eidx[EE + e]; }
    else if (e < ENTOT) { sj = di = e - EE; }
    else                { sj = di = 0; }
    sdst[tid] = di; ssrc[tid] = sj;
  }
  if (tid < CH) { sB1[tid] = b1x[tid]; sS1[tid] = st[S1X + tid]; sT1[tid] = st[T1X + tid]; }
  __syncthreads();
  for (int v = tid; v < 16 * CH; v += 256) {
    int row = v >> 7, ch = v & 127;
    float h = Ax[(size_t)sdst[row] * CH + ch] + Bx[(size_t)ssrc[row] * CH + ch] + sB1[ch];
    h = fmaxf(h, 0.f);
    nf[afrag_idx(row, ch)] = f2bf(sS1[ch] * h + sT1[ch]);
  }
  __syncthreads();
  int w = tid >> 5, lane = tid & 31;
  v8f acc = {0.f, 0.f, 0.f, 0.f, 0.f, 0.f, 0.f, 0.f};
  #pragma unroll
  for (int s = 0; s < 4; s++) {
    v16bf a = *reinterpret_cast<const v16bf*>(&nf[(s * 32 + lane) * 16]);
    v16bf b = *reinterpret_cast<const v16bf*>(w2f + (size_t)((w * 4 + s) * 32 + lane) * 8);
    acc = __builtin_amdgcn_wmma_f32_16x16x32_bf16(false, a, false, b, (short)0, acc, false, false);
  }
  int col = w * 16 + (lane & 15);
  int rw0 = (lane >> 4) << 3;                  // first D-row this lane owns
  float bias = b2x[col], s = 0.f, q = 0.f;
  if (base + 16 <= ENTOT) {                    // full tile fast path
    #pragma unroll
    for (int r = 0; r < 8; r++) {
      float val = fmaxf(acc[r] + bias, 0.f);
      // >=0 floats: IEEE order == int order; buffers pre-zeroed.
      atomicMax((int*)&maxx[(size_t)sdst[rw0 + r] * CH + col], __float_as_int(val));
      s += val; q += val * val;
    }
  } else {
    #pragma unroll
    for (int r = 0; r < 8; r++) {
      float val = fmaxf(acc[r] + bias, 0.f);
      if (base + rw0 + r < ENTOT) {
        atomicMax((int*)&maxx[(size_t)sdst[rw0 + r] * CH + col], __float_as_int(val));
        s += val; q += val * val;
      }
    }
  }
  s += __shfl_xor(s, 16, 32);
  q += __shfl_xor(q, 16, 32);
  if (lane < 16) { atomicAdd(&st[SUM2X + col], s); atomicAdd(&st[SQ2X + col], q); }
}

// Pos path layer-2 (16x16 dense, VALU) + stats + segment-max.
__global__ void k_edge_pos(const int* eidx, const float* PA, const float* PB,
                           const float* b1p, const float* Wp2, const float* b2p,
                           float* st, float* maxp) {
  __shared__ float sW[256], sb2[16], sS1[16], sT1[16], sb1[16], sAcc[16], sAq[16];
  int tid = threadIdx.x;
  sW[tid] = Wp2[tid];
  if (tid < 16) {
    sb2[tid] = b2p[tid]; sS1[tid] = st[S1P + tid]; sT1[tid] = st[T1P + tid];
    sb1[tid] = b1p[tid]; sAcc[tid] = 0.f; sAq[tid] = 0.f;
  }
  __syncthreads();
  int e = blockIdx.x * 256 + tid;
  if (e < ENTOT) {
    int sj, di;
    if (e < EE) { sj = eidx[e]; di = eidx[EE + e]; } else { sj = di = e - EE; }
    float n1[16];
    #pragma unroll
    for (int k = 0; k < 16; k++) {
      float h = PA[di * PF + k] + PB[sj * PF + k] + sb1[k];
      h = fmaxf(h, 0.f);
      n1[k] = sS1[k] * h + sT1[k];
    }
    #pragma unroll
    for (int c = 0; c < 16; c++) {
      float a = sb2[c];
      #pragma unroll
      for (int k = 0; k < 16; k++) a += n1[k] * sW[k * 16 + c];
      a = fmaxf(a, 0.f);
      atomicMax((int*)&maxp[(size_t)di * PF + c], __float_as_int(a));
      atomicAdd(&sAcc[c], a); atomicAdd(&sAq[c], a * a);   // ds_add_f32
    }
  }
  __syncthreads();
  if (tid < 16) { atomicAdd(&st[SUM2P + tid], sAcc[tid]); atomicAdd(&st[SQ2P + tid], sAq[tid]); }
}

// Finalize BN2 -> combined concat affine (a>0 since g=1, so max commutes).
__global__ void k_fin_bn2(const float* st, const float* g2x, const float* be2x,
                          const float* g2p, const float* be2p,
                          float* acat, float* ccat, int goff) {
  int c = threadIdx.x;
  const float inv_n = 1.f / (float)ENTOT;
  if (c < CH) {
    float m = st[SUM2X + c] * inv_n;
    float v = fmaxf(st[SQ2X + c] * inv_n - m * m, 0.f);
    float a = g2x[c] * rsqrtf(v + BNEPS);
    acat[goff + c] = a; ccat[goff + c] = be2x[c] - a * m;
  } else if (c < CH + PF) {
    int cc = c - CH;
    float m = st[SUM2P + cc] * inv_n;
    float v = fmaxf(st[SQ2P + cc] * inv_n - m * m, 0.f);
    float a = g2p[cc] * rsqrtf(v + BNEPS);
    acat[goff + CH + cc] = a; ccat[goff + CH + cc] = be2p[cc] - a * m;
  }
}

// Final: z = relu(concat(bn-affine(max bufs)) @ W + b), plus BN stats over N.
__global__ void k_final_gemm(const float* maxx0, const float* maxp0,
                             const float* maxx1, const float* maxp1,
                             const float* acat, const float* ccat,
                             const unsigned int* woutf, const float* bout,
                             float* z, float* fst) {
  __shared__ __align__(32) unsigned short af[9 * 32 * 16];
  __shared__ float sA[DCAT], sC[DCAT];
  int tid = threadIdx.x, base = blockIdx.x * 16;
  for (int c = tid; c < DCAT; c += 256) { sA[c] = acat[c]; sC[c] = ccat[c]; }
  __syncthreads();
  for (int v = tid; v < 16 * DCAT; v += 256) {
    int ch = v >> 4, row = v & 15;
    int node = base + row;
    float out = 0.f;
    if (node < NN) {
      int cl = (ch < 144) ? ch : ch - 144;
      const float* mx = (ch < 144) ? maxx0 : maxx1;
      const float* mp = (ch < 144) ? maxp0 : maxp1;
      float raw = (cl < CH) ? mx[(size_t)node * CH + cl] : mp[(size_t)node * PF + (cl - CH)];
      out = sA[ch] * raw + sC[ch];
    }
    af[afrag_idx(row, ch)] = f2bf(out);
  }
  __syncthreads();
  int w = tid >> 5, lane = tid & 31;
  bool full = (base + 16 <= NN);
  for (int tt = 0; tt < 2; tt++) {
    int t = w + tt * 8;
    v8f acc = {0.f, 0.f, 0.f, 0.f, 0.f, 0.f, 0.f, 0.f};
    #pragma unroll
    for (int s = 0; s < 9; s++) {
      v16bf a = *reinterpret_cast<const v16bf*>(&af[(s * 32 + lane) * 16]);
      v16bf b = *reinterpret_cast<const v16bf*>(woutf + (size_t)((t * 9 + s) * 32 + lane) * 8);
      acc = __builtin_amdgcn_wmma_f32_16x16x32_bf16(false, a, false, b, (short)0, acc, false, false);
    }
    int col  = t * 16 + (lane & 15);
    int row0 = base + ((lane >> 4) << 3);
    float bias = bout[col], s = 0.f, q = 0.f;
    float* p = z + (size_t)row0 * CO + col;
    if (full) {
      #pragma unroll
      for (int r = 0; r < 8; r++) {
        float val = fmaxf(acc[r] + bias, 0.f);
        p[(size_t)r * CO] = val; s += val; q += val * val;
      }
    } else {
      #pragma unroll
      for (int r = 0; r < 8; r++) {
        float val = fmaxf(acc[r] + bias, 0.f);
        if (row0 + r < NN) { p[(size_t)r * CO] = val; s += val; q += val * val; }
      }
    }
    s += __shfl_xor(s, 16, 32);
    q += __shfl_xor(q, 16, 32);
    if (lane < 16) { atomicAdd(&fst[SUMZ + col], s); atomicAdd(&fst[SQZ + col], q); }
  }
}

__global__ void k_fin_bnout(float* fst, const float* gout, const float* beout) {
  int c = threadIdx.x;
  const float inv_n = 1.f / (float)NN;
  float m = fst[SUMZ + c] * inv_n;
  float v = fmaxf(fst[SQZ + c] * inv_n - m * m, 0.f);
  float a = gout[c] * rsqrtf(v + BNEPS);
  fst[AOUT + c] = a; fst[COUTO + c] = beout[c] - a * m;
}

__global__ void k_apply(const float* z, const float* fst, float* out) {
  int id = blockIdx.x * 256 + threadIdx.x;
  if (id >= NN * CO) return;
  int c = id & 255;
  out[id] = fst[AOUT + c] * z[id] + fst[COUTO + c];
}

// ---------------------------------------------------------------------------
extern "C" void kernel_launch(void* const* d_in, const int* in_sizes, int n_in,
                              void* d_out, int out_size, void* d_ws, size_t ws_size,
                              hipStream_t stream) {
  (void)in_sizes; (void)n_in; (void)out_size; (void)ws_size;
  const float* pos = (const float*)d_in[0];
  const float* x   = (const float*)d_in[1];
  const int* eidx[2] = { (const int*)d_in[2], (const int*)d_in[3] };
  const float *xW1[2], *xb1[2], *xg1[2], *xbe1[2], *xW2[2], *xb2[2], *xg2[2], *xbe2[2];
  const float *pW1[2], *pb1[2], *pg1[2], *pbe1[2], *pW2[2], *pb2[2], *pg2[2], *pbe2[2];
  for (int g = 0; g < 2; g++) {
    int b = 4 + g * 16;
    xW1[g]=(const float*)d_in[b+0];  xb1[g]=(const float*)d_in[b+1];
    xg1[g]=(const float*)d_in[b+2];  xbe1[g]=(const float*)d_in[b+3];
    xW2[g]=(const float*)d_in[b+4];  xb2[g]=(const float*)d_in[b+5];
    xg2[g]=(const float*)d_in[b+6];  xbe2[g]=(const float*)d_in[b+7];
    pW1[g]=(const float*)d_in[b+8];  pb1[g]=(const float*)d_in[b+9];
    pg1[g]=(const float*)d_in[b+10]; pbe1[g]=(const float*)d_in[b+11];
    pW2[g]=(const float*)d_in[b+12]; pb2[g]=(const float*)d_in[b+13];
    pg2[g]=(const float*)d_in[b+14]; pbe2[g]=(const float*)d_in[b+15];
  }
  const float* Wout  = (const float*)d_in[36];
  const float* bout  = (const float*)d_in[37];
  const float* gout  = (const float*)d_in[38];
  const float* beout = (const float*)d_in[39];

  char* wsb = (char*)d_ws;
  size_t off = 0;
  auto alloc = [&](size_t bytes) -> void* {
    off = (off + 255) & ~(size_t)255;
    void* p = wsb + off;
    off += bytes;
    return p;
  };

  // --- zero-on-every-call region (contiguous): max bufs + stat blocks ---
  off = (off + 255) & ~(size_t)255;
  size_t zstart = off;
  float* maxx[2]; float* maxp[2]; float* st[2];
  maxx[0] = (float*)alloc((size_t)NN * CH * 4);
  maxp[0] = (float*)alloc((size_t)NN * PF * 4);
  maxx[1] = (float*)alloc((size_t)NN * CH * 4);
  maxp[1] = (float*)alloc((size_t)NN * PF * 4);
  st[0]   = (float*)alloc(STATSZ * 4);
  st[1]   = (float*)alloc(STATSZ * 4);
  float* fst = (float*)alloc(FSTSZ * 4);
  size_t zbytes = ((off + 255) & ~(size_t)255) - zstart;
  float* zptr = (float*)(wsb + zstart);

  // --- persistent-within-call scratch ---
  float* Ax[2]; float* Bx[2]; float* PA[2]; float* PB[2];
  unsigned int* w1df[2]; unsigned int* w1bf[2]; unsigned int* w2f[2];
  for (int g = 0; g < 2; g++) {
    Ax[g]  = (float*)alloc((size_t)NN * CH * 4);
    Bx[g]  = (float*)alloc((size_t)NN * CH * 4);
    PA[g]  = (float*)alloc((size_t)NN * PF * 4);
    PB[g]  = (float*)alloc((size_t)NN * PF * 4);
    w1df[g] = (unsigned int*)alloc(8 * 2 * 32 * 8 * 4);
    w1bf[g] = (unsigned int*)alloc(8 * 2 * 32 * 8 * 4);
    w2f[g]  = (unsigned int*)alloc(8 * 4 * 32 * 8 * 4);
  }
  unsigned short* xbf = (unsigned short*)alloc((size_t)NN * CIN * 2);
  unsigned int* woutf = (unsigned int*)alloc(16 * 9 * 32 * 8 * 4);
  float* acat = (float*)alloc(DCAT * 4);
  float* ccat = (float*)alloc(DCAT * 4);
  float* z    = (float*)alloc((size_t)NN * CO * 4);

  int zf = (int)(zbytes / 4);
  k_zero<<<(zf + 255) / 256, 256, 0, stream>>>(zptr, zf);
  k_cvt_bf16<<<(NN * CIN + 255) / 256, 256, 0, stream>>>(x, xbf, NN * CIN);

  for (int g = 0; g < 2; g++) {
    k_pack_w1<<<16, 256, 0, stream>>>(xW1[g], w1df[g], w1bf[g]);
    k_pack_gen<<<32, 256, 0, stream>>>(xW2[g], w2f[g], 128, 128, 8, 4);
  }
  k_pack_gen<<<144, 256, 0, stream>>>(Wout, woutf, DCAT, CO, 16, 9);

  const int ntile = (NN + 15) / 16;        // 1563
  const int etile = (ENTOT + 15) / 16;     // 26563
  for (int g = 0; g < 2; g++) {
    k_node_gemm<<<ntile, 256, 0, stream>>>(xbf, w1df[g], Ax[g]);
    k_node_gemm<<<ntile, 256, 0, stream>>>(xbf, w1bf[g], Bx[g]);
    k_pos_pre<<<(NN * PF + 255) / 256, 256, 0, stream>>>(pos, pW1[g], PA[g], PB[g]);
    k_stats1<<<(ENTOT + 63) / 64, 256, 0, stream>>>(eidx[g], Ax[g], Bx[g], xb1[g],
                                                    PA[g], PB[g], pb1[g], st[g]);
    k_fin_bn1<<<1, 256, 0, stream>>>(st[g], xg1[g], xbe1[g], pg1[g], pbe1[g]);
    k_edge_gemm<<<etile, 256, 0, stream>>>(eidx[g], Ax[g], Bx[g], xb1[g], xb2[g],
                                           w2f[g], st[g], maxx[g]);
    k_edge_pos<<<(ENTOT + 255) / 256, 256, 0, stream>>>(eidx[g], PA[g], PB[g], pb1[g],
                                                        pW2[g], pb2[g], st[g], maxp[g]);
    k_fin_bn2<<<1, 256, 0, stream>>>(st[g], xg2[g], xbe2[g], pg2[g], pbe2[g],
                                     acat, ccat, g * 144);
  }
  k_final_gemm<<<ntile, 256, 0, stream>>>(maxx[0], maxp[0], maxx[1], maxp[1],
                                          acat, ccat, woutf, bout, z, fst);
  k_fin_bnout<<<1, 256, 0, stream>>>(fst, gout, beout);
  k_apply<<<(NN * CO + 255) / 256, 256, 0, stream>>>(z, fst, (float*)d_out);
}
